// InstantPolicy_18640158064687
// MI455X (gfx1250) — compile-verified, matrix operand order checked by reading, not run on hardware
//
#include <hip/hip_runtime.h>
#include <hip/hip_fp16.h>

// ---------------------------------------------------------------------------
// CDNA5 (gfx1250) implementation of a 2-layer heterogeneous graph attention
// network.  All dense math runs through v_wmma_f32_16x16x32_f16 (wave32 WMMA).
// The sincos edge embedding AND the e@We projection AND the attention score
// are fused into one kernel so the [E,512] embedding / key matrices are never
// materialized in HBM.  Edge tile = 32 edges/block so each wave reuses its B
// (WeT) registers across 2 A sub-tiles (halves L2 traffic on the hot loop).
// ---------------------------------------------------------------------------

typedef __attribute__((ext_vector_type(16))) _Float16 v16h;
typedef __attribute__((ext_vector_type(8)))  float    v8f;

union HVec { v16h v; uint4 u[2]; };

#define NHID 512
#define NHEADS 8
#define HDIM 64
#define ETILE 32   // edges per block in edge_scores_k

// --------------------------- utility kernels -------------------------------

__global__ void to_half_k(const float* __restrict__ x, __half* __restrict__ y, int n) {
    int i = blockIdx.x * 256 + threadIdx.x;
    if (i < n) y[i] = __float2half(x[i]);
}

__global__ void fill_f32_k(float* __restrict__ p, float v, int n) {
    int i = blockIdx.x * 256 + threadIdx.x;
    if (i < n) p[i] = v;
}

__global__ void fill_u32_k(unsigned* __restrict__ p, unsigned v, int n) {
    int i = blockIdx.x * 256 + threadIdx.x;
    if (i < n) p[i] = v;
}

// Wt[n][k] = (half) W[k][n]  (512x512), LDS-tiled transpose.
// grid (16,16), block (32,8)
__global__ void w_transpose_h_k(const float* __restrict__ W, __half* __restrict__ Wt) {
    __shared__ float tile[32][33];
    int bx = blockIdx.x * 32, by = blockIdx.y * 32;
    #pragma unroll
    for (int i = 0; i < 32; i += 8)
        tile[threadIdx.y + i][threadIdx.x] = W[(size_t)(by + threadIdx.y + i) * NHID + bx + threadIdx.x];
    __syncthreads();
    #pragma unroll
    for (int i = 0; i < 32; i += 8)
        Wt[(size_t)(bx + threadIdx.y + i) * NHID + by + threadIdx.x] =
            __float2half(tile[threadIdx.x][threadIdx.y + i]);
}

// --------------------------- WMMA GEMM -------------------------------------
// C[M,512] (f32) = A[M,512] (f16, row major) * W[512,512], where
// Bt[n][k] = W[k][n] in f16.  Block = 256 threads = 8 waves; wave w owns
// columns [64w, 64w+64).
__global__ __launch_bounds__(256) void gemm_h16_k(const __half* __restrict__ A,
                                                  const __half* __restrict__ Bt,
                                                  float* __restrict__ C, int M) {
    const int t = threadIdx.x;
    const int w = t >> 5, lane = t & 31;
    const int n16 = lane & 15, hi = lane >> 4;
    const int mrow = blockIdx.x * 16 + n16;
    const int ma = (mrow < M) ? mrow : (M - 1);
    const int nbase = w * 64;
    const __half* Arow = A + (size_t)ma * NHID;

    v8f acc[4];
    #pragma unroll
    for (int q = 0; q < 4; ++q)
        #pragma unroll
        for (int i = 0; i < 8; ++i) acc[q][i] = 0.0f;

    for (int kk = 0; kk < NHID; kk += 32) {
        const int kb = kk + hi * 8;
        HVec a;
        a.u[0] = *(const uint4*)(Arow + kb);
        a.u[1] = *(const uint4*)(Arow + kb + 16);
        #pragma unroll
        for (int tt = 0; tt < 4; ++tt) {
            const int n = nbase + tt * 16 + n16;
            const __half* Brow = Bt + (size_t)n * NHID;
            HVec b;
            b.u[0] = *(const uint4*)(Brow + kb);
            b.u[1] = *(const uint4*)(Brow + kb + 16);
            acc[tt] = __builtin_amdgcn_wmma_f32_16x16x32_f16(
                false, a.v, false, b.v, (short)0, acc[tt], false, false);
        }
    }
    #pragma unroll
    for (int tt = 0; tt < 4; ++tt)
        #pragma unroll
        for (int r = 0; r < 8; ++r) {
            const int m = blockIdx.x * 16 + r + 8 * hi;
            const int n = nbase + tt * 16 + n16;
            if (m < M) C[(size_t)m * NHID + n] = acc[tt][r];
        }
}

// ------------------ fused edge-emb + e@We + score kernel --------------------
// For a tile of 32 edges:
//   1) build sincos embedding e_tile[32][512] (f16) in LDS
//   2) wave w computes ke = e_tile @ We[:, 64w:64w+64] via WMMA,
//      reusing the 4 B register tiles across 2 A sub-tiles (16 edges each)
//   3) s[e, w] = sum_d q[dst,64w+d] * (k[src,64w+d] + ke[d]) / 8
// grid = E/32, block = 256.
__global__ __launch_bounds__(256) void edge_scores_k(
        const int* __restrict__ ei_src, const int* __restrict__ ei_dst,
        const float* __restrict__ pos_src, const float* __restrict__ pos_dst,
        const __half* __restrict__ WeT,
        const float* __restrict__ qbuf, const float* __restrict__ kbuf,
        float* __restrict__ sbuf) {
    __shared__ __align__(16) __half et[ETILE][NHID];
    __shared__ float freqs[128];
    __shared__ float dif[ETILE][2];
    __shared__ int ssrc[ETILE], sdst[ETILE];

    const int t = threadIdx.x;
    const int ebase = blockIdx.x * ETILE;

    if (t < 128) freqs[t] = exp2f((float)t) * 3.14159265358979323846f; // freqs[127] -> +inf (matches fp32 ref)
    if (t < ETILE) { ssrc[t] = ei_src[ebase + t]; sdst[t] = ei_dst[ebase + t]; }
    __syncthreads();

    if (t < 2 * ETILE) {
        const int me = t >> 1, p = t & 1;
        float d = pos_dst[(size_t)sdst[me] * 2 + p] - pos_src[(size_t)ssrc[me] * 2 + p];
        dif[me][p] = fminf(fmaxf(d, -1000.0f), 1000.0f);
    }
    __syncthreads();

    {   // embedding generation: thread t covers edge t>>3, 32 freq-slots
        const int me = t >> 3, j = t & 7;
        const float d0 = dif[me][0], d1 = dif[me][1];
        #pragma unroll
        for (int i = 0; i < 32; ++i) {
            const int fp = j * 32 + i;            // 0..255
            const int p = fp >> 7, d = fp & 127;
            const float ang = (p ? d1 : d0) * freqs[d];
            float sv, cv;
            sincosf(ang, &sv, &cv);
            const int col = (p << 8) + (d << 1);  // layout: [p, d, {sin,cos}]
            et[me][col]     = __float2half(sv);
            et[me][col + 1] = __float2half(cv);
        }
    }
    __syncthreads();

    const int w = t >> 5, lane = t & 31;
    const int n16 = lane & 15, hi = lane >> 4;
    const __half* Brow0 = WeT + (size_t)(w * 64 + n16) * NHID;

    v8f acc[2][4];
    #pragma unroll
    for (int es = 0; es < 2; ++es)
        #pragma unroll
        for (int q = 0; q < 4; ++q)
            #pragma unroll
            for (int i = 0; i < 8; ++i) acc[es][q][i] = 0.0f;

    for (int kk = 0; kk < NHID; kk += 32) {
        const int kb = kk + hi * 8;
        if (kk + 32 < NHID)                                   // pull next k-slab toward WGP
            __builtin_prefetch(Brow0 + kb + 32, 0, 0);        // -> global_prefetch_b8
        HVec b[4];
        #pragma unroll
        for (int tt = 0; tt < 4; ++tt) {
            const __half* Brow = Brow0 + (size_t)(tt * 16) * NHID;
            b[tt].u[0] = *(const uint4*)(Brow + kb);
            b[tt].u[1] = *(const uint4*)(Brow + kb + 16);
        }
        #pragma unroll
        for (int es = 0; es < 2; ++es) {
            HVec a;
            a.u[0] = *(const uint4*)(&et[es * 16 + n16][kb]);
            a.u[1] = *(const uint4*)(&et[es * 16 + n16][kb + 16]);
            #pragma unroll
            for (int tt = 0; tt < 4; ++tt)
                acc[es][tt] = __builtin_amdgcn_wmma_f32_16x16x32_f16(
                    false, a.v, false, b[tt].v, (short)0, acc[es][tt], false, false);
        }
    }

    // score reduction: this wave handles head h == w (64 columns)
    #pragma unroll
    for (int es = 0; es < 2; ++es) {
        #pragma unroll
        for (int r = 0; r < 8; ++r) {
            const int me = es * 16 + r + 8 * hi;
            const size_t qrow = (size_t)sdst[me] * NHID;
            const size_t krow = (size_t)ssrc[me] * NHID;
            float part = 0.0f;
            #pragma unroll
            for (int tt = 0; tt < 4; ++tt) {
                const int col = w * 64 + tt * 16 + n16;
                part += qbuf[qrow + col] * (kbuf[krow + col] + acc[es][tt][r]);
            }
            part += __shfl_xor(part, 1, 32);
            part += __shfl_xor(part, 2, 32);
            part += __shfl_xor(part, 4, 32);
            part += __shfl_xor(part, 8, 32);
            if (n16 == 0)
                sbuf[(size_t)(ebase + me) * NHEADS + w] = part * 0.125f; // 1/sqrt(64)
        }
    }
}

// --------------------------- segment softmax -------------------------------

__device__ __forceinline__ unsigned fenc(float x) {
    unsigned u = __float_as_uint(x);
    return (u & 0x80000000u) ? ~u : (u | 0x80000000u);
}
__device__ __forceinline__ float fdec(unsigned e) {
    unsigned u = (e & 0x80000000u) ? (e & 0x7FFFFFFFu) : ~e;
    return __uint_as_float(u);
}

__global__ void seg_max_k(const float* __restrict__ sbuf, const int* __restrict__ ei_dst,
                          unsigned* __restrict__ menc, int E8) {
    int i = blockIdx.x * 256 + threadIdx.x;
    if (i >= E8) return;
    int e = i >> 3, h = i & 7;
    atomicMax(&menc[(size_t)ei_dst[e] * NHEADS + h], fenc(sbuf[i]));
}

__global__ void max_finish_k(const unsigned* __restrict__ menc, float* __restrict__ mval, int n) {
    int i = blockIdx.x * 256 + threadIdx.x;
    if (i >= n) return;
    float m = fdec(menc[i]);
    bool fin = ((__float_as_uint(m) & 0x7F800000u) != 0x7F800000u);
    mval[i] = fin ? m : 0.0f;    // where(isfinite(m), m, 0)
}

__global__ void exp_den_k(float* __restrict__ sbuf, const int* __restrict__ ei_dst,
                          const float* __restrict__ mval, float* __restrict__ den, int E8) {
    int i = blockIdx.x * 256 + threadIdx.x;
    if (i >= E8) return;
    int e = i >> 3, h = i & 7;
    int d = ei_dst[e] * NHEADS + h;
    float ex = expf(sbuf[i] - mval[d]);
    sbuf[i] = ex;
    atomicAdd(&den[d], ex);
}

// message scatter: agg[dst, col] += (ex/(den+1e-9)) * v[src, col]
__global__ void scatter_msg_k(const float* __restrict__ sbuf, const float* __restrict__ den,
                              const int* __restrict__ ei_src, const int* __restrict__ ei_dst,
                              const float* __restrict__ vbuf, float* __restrict__ agg,
                              long long total) {
    long long i = (long long)blockIdx.x * 256 + threadIdx.x;
    if (i >= total) return;
    int e = (int)(i >> 9);
    int col = (int)(i & 511);
    int h = col >> 6;
    int s = ei_src[e], d = ei_dst[e];
    float a = sbuf[(size_t)e * NHEADS + h] / (den[(size_t)d * NHEADS + h] + 1e-9f);
    atomicAdd(&agg[(size_t)d * NHID + col], a * vbuf[(size_t)s * NHID + col]);
}

// ------------------------ layernorm + residual -----------------------------
// out = LN(x)*g + b + resid ; optionally also writes f16 copy for next-layer
// GEMM input.  One block (256 threads) per row of 512.
__global__ __launch_bounds__(256) void ln_res_k(const float* __restrict__ x,
                                                const float* __restrict__ g,
                                                const float* __restrict__ b,
                                                const float* __restrict__ resid,
                                                float* __restrict__ out,
                                                __half* __restrict__ outh, int N) {
    const int row = blockIdx.x;
    if (row >= N) return;
    const int t = threadIdx.x;
    const size_t base = (size_t)row * NHID;
    float v0 = x[base + t], v1 = x[base + t + 256];

    __shared__ float wsum[8];
    __shared__ float stats[2];

    float s = v0 + v1;
    for (int m = 1; m < 32; m <<= 1) s += __shfl_xor(s, m, 32);
    if ((t & 31) == 0) wsum[t >> 5] = s;
    __syncthreads();
    if (t == 0) {
        float tot = 0.0f;
        #pragma unroll
        for (int i = 0; i < 8; ++i) tot += wsum[i];
        stats[0] = tot * (1.0f / NHID);
    }
    __syncthreads();
    const float mu = stats[0];
    const float d0 = v0 - mu, d1 = v1 - mu;
    float q = d0 * d0 + d1 * d1;
    __syncthreads();
    for (int m = 1; m < 32; m <<= 1) q += __shfl_xor(q, m, 32);
    if ((t & 31) == 0) wsum[t >> 5] = q;
    __syncthreads();
    if (t == 0) {
        float tot = 0.0f;
        #pragma unroll
        for (int i = 0; i < 8; ++i) tot += wsum[i];
        stats[1] = rsqrtf(tot * (1.0f / NHID) + 1e-5f);
    }
    __syncthreads();
    const float rs = stats[1];
    float y0 = d0 * rs * g[t]       + b[t]       + resid[base + t];
    float y1 = d1 * rs * g[t + 256] + b[t + 256] + resid[base + t + 256];
    out[base + t] = y0;
    out[base + t + 256] = y1;
    if (outh) {
        outh[base + t] = __float2half(y0);
        outh[base + t + 256] = __float2half(y1);
    }
}

// ------------------------------- host side ---------------------------------

extern "C" void kernel_launch(void* const* d_in, const int* in_sizes, int n_in,
                              void* d_out, int out_size, void* d_ws, size_t ws_size,
                              hipStream_t stream) {
    (void)in_sizes; (void)n_in; (void)out_size; (void)ws_size;

    const int NA = 512, NO = 10000;
    const int EAA = 8192, EOA = 65536, EOO = 131072;

    // ---- input layout (params pytree flattened with sorted dict keys) ----
    // per layer (18 leaves): 0 aa.We 1 aa.Wk 2 aa.Wq 3 aa.Wv
    //   4 agent.Wo 5 agent.b 6 agent.g
    //   7 oa.We 8 oa.Wk 9 oa.Wq 10 oa.Wv
    //   11 object.Wo 12 object.b 13 object.g
    //   14 oo.We 15 oo.Wk 16 oo.Wq 17 oo.Wv
    const float* x_agent   = (const float*)d_in[36];
    const float* x_object  = (const float*)d_in[37];
    const float* pos_agent = (const float*)d_in[38];
    const float* pos_object= (const float*)d_in[39];
    const int*   ei_aa     = (const int*)d_in[40];
    const int*   ei_oa     = (const int*)d_in[41];
    const int*   ei_oo     = (const int*)d_in[42];

    // ---- workspace carve ----
    char* wp = (char*)d_ws;
    auto alloc = [&](size_t bytes) -> char* {
        char* p = wp;
        wp += (bytes + 255) & ~(size_t)255;
        return p;
    };
    const size_t WSZ = (size_t)NHID * NHID; // 262144
    __half*   wt     = (__half*)alloc(28 * WSZ * sizeof(__half));  // 14 per layer, transposed f16
    __half*   xh_a   = (__half*)alloc((size_t)NA * NHID * sizeof(__half));
    __half*   xh_o   = (__half*)alloc((size_t)NO * NHID * sizeof(__half));
    float*    z_a    = (float*)alloc((size_t)NA * NHID * sizeof(float));
    float*    z_o    = (float*)alloc((size_t)NO * NHID * sizeof(float));
    __half*   zh_a   = (__half*)alloc((size_t)NA * NHID * sizeof(__half));
    __half*   zh_o   = (__half*)alloc((size_t)NO * NHID * sizeof(__half));
    __half*   aggh   = (__half*)alloc((size_t)NO * NHID * sizeof(__half));
    float*    qb     = (float*)alloc((size_t)NO * NHID * sizeof(float));
    float*    kb     = (float*)alloc((size_t)NO * NHID * sizeof(float));
    float*    vb     = (float*)alloc((size_t)NO * NHID * sizeof(float));
    float*    sb     = (float*)alloc((size_t)EOO * NHEADS * sizeof(float));
    unsigned* menc   = (unsigned*)alloc((size_t)NO * NHEADS * sizeof(unsigned));
    float*    mval   = (float*)alloc((size_t)NO * NHEADS * sizeof(float));
    float*    den    = (float*)alloc((size_t)NO * NHEADS * sizeof(float));
    float*    agg_a  = (float*)alloc((size_t)NA * NHID * sizeof(float));
    float*    agg_o  = (float*)alloc((size_t)NO * NHID * sizeof(float));

    // ---- one-time staging: f16 inputs + transposed f16 weights ----
    to_half_k<<<(NA * NHID + 255) / 256, 256, 0, stream>>>(x_agent, xh_a, NA * NHID);
    to_half_k<<<(NO * NHID + 255) / 256, 256, 0, stream>>>(x_object, xh_o, NO * NHID);

    // wt slots per layer: 0 aa.We 1 aa.Wk 2 aa.Wq 3 aa.Wv 4 agent.Wo
    //                     5 oa.We 6 oa.Wk 7 oa.Wq 8 oa.Wv 9 object.Wo
    //                     10 oo.We 11 oo.Wk 12 oo.Wq 13 oo.Wv
    const int widx[14] = {0, 1, 2, 3, 4, 7, 8, 9, 10, 11, 14, 15, 16, 17};
    for (int l = 0; l < 2; ++l)
        for (int j = 0; j < 14; ++j)
            w_transpose_h_k<<<dim3(16, 16), dim3(32, 8), 0, stream>>>(
                (const float*)d_in[l * 18 + widx[j]], wt + (size_t)(l * 14 + j) * WSZ);

    auto gemm = [&](const __half* A, const __half* Bt, float* C, int M) {
        gemm_h16_k<<<dim3((M + 15) / 16), dim3(256), 0, stream>>>(A, Bt, C, M);
    };
    auto fillf = [&](float* p, float v, int n) {
        fill_f32_k<<<(n + 255) / 256, 256, 0, stream>>>(p, v, n);
    };

    float* y_agent  = (float*)d_out;                        // [512,512]
    float* y_object = (float*)d_out + (size_t)NA * NHID;    // [10000,512]

    for (int l = 0; l < 2; ++l) {
        const __half* fa = (l == 0) ? xh_a : zh_a;
        const __half* fo = (l == 0) ? xh_o : zh_o;
        const float*  ra = (l == 0) ? x_agent : z_a;
        const float*  ro = (l == 0) ? x_object : z_o;
        __half* wl = wt + (size_t)l * 14 * WSZ;

        fillf(agg_a, 0.0f, NA * NHID);
        fillf(agg_o, 0.0f, NO * NHID);

        struct ETcfg {
            const int* ei; int E;
            const __half* fs; const __half* fd;
            const float* ps; const float* pd;
            int Ns, Nd;
            int sWe, sWk, sWq, sWv;
            float* agg;
        } ets[3] = {
            { ei_aa, EAA, fa, fa, pos_agent,  pos_agent, NA, NA, 0, 1, 2, 3, agg_a },
            { ei_oa, EOA, fo, fa, pos_object, pos_agent, NO, NA, 5, 6, 7, 8, agg_a },
            { ei_oo, EOO, fo, fo, pos_object, pos_object, NO, NO, 10, 11, 12, 13, agg_o },
        };

        for (int e = 0; e < 3; ++e) {
            const ETcfg& c = ets[e];
            const int* src = c.ei;
            const int* dst = c.ei + c.E;
            const __half* WeT = wl + (size_t)c.sWe * WSZ;
            const __half* WkT = wl + (size_t)c.sWk * WSZ;
            const __half* WqT = wl + (size_t)c.sWq * WSZ;
            const __half* WvT = wl + (size_t)c.sWv * WSZ;

            gemm(c.fd, WqT, qb, c.Nd);   // q   = feats[dt] @ Wq
            gemm(c.fs, WkT, kb, c.Ns);   // k_s = feats[st] @ Wk
            gemm(c.fs, WvT, vb, c.Ns);   // v   = feats[st] @ Wv

            fill_u32_k<<<(c.Nd * NHEADS + 255) / 256, 256, 0, stream>>>(
                menc, 0x007FFFFFu /* enc(-inf) */, c.Nd * NHEADS);
            fillf(den, 0.0f, c.Nd * NHEADS);

            edge_scores_k<<<c.E / ETILE, 256, 0, stream>>>(
                src, dst, c.ps, c.pd, WeT, qb, kb, sb);

            const int E8 = c.E * NHEADS;
            seg_max_k<<<(E8 + 255) / 256, 256, 0, stream>>>(sb, dst, menc, E8);
            max_finish_k<<<(c.Nd * NHEADS + 255) / 256, 256, 0, stream>>>(menc, mval, c.Nd * NHEADS);
            exp_den_k<<<(E8 + 255) / 256, 256, 0, stream>>>(sb, dst, mval, den, E8);

            long long total = (long long)c.E * NHID;
            scatter_msg_k<<<(unsigned)((total + 255) / 256), 256, 0, stream>>>(
                sb, den, src, dst, vb, c.agg, total);
        }

        // ---- output projection + residual LayerNorm per node type ----
        // agent
        to_half_k<<<(NA * NHID + 255) / 256, 256, 0, stream>>>(agg_a, aggh, NA * NHID);
        gemm(aggh, wl + (size_t)4 * WSZ, kb, NA);           // agg_a @ agent.Wo -> kb
        ln_res_k<<<NA, 256, 0, stream>>>(
            kb, (const float*)d_in[l * 18 + 6], (const float*)d_in[l * 18 + 5], ra,
            (l == 0) ? z_a : y_agent, (l == 0) ? zh_a : (__half*)nullptr, NA);
        // object
        to_half_k<<<(NO * NHID + 255) / 256, 256, 0, stream>>>(agg_o, aggh, NO * NHID);
        gemm(aggh, wl + (size_t)9 * WSZ, kb, NO);           // agg_o @ object.Wo -> kb
        ln_res_k<<<NO, 256, 0, stream>>>(
            kb, (const float*)d_in[l * 18 + 13], (const float*)d_in[l * 18 + 12], ro,
            (l == 0) ? z_o : y_object, (l == 0) ? zh_o : (__half*)nullptr, NO);
    }
}
